// DAHead_79405355369273
// MI455X (gfx1250) — matrix operand order repaired and checked by reading, not compiled, not run on hardware
//
#include <hip/hip_runtime.h>

// ---------------------------------------------------------------------------
// Types for gfx1250 WMMA
// ---------------------------------------------------------------------------
typedef __bf16 bf16;
typedef __attribute__((ext_vector_type(16))) __bf16 v16bf;
typedef __attribute__((ext_vector_type(8)))  float   v8f;

#define WMMA_BF16(A_, B_, C_) \
  __builtin_amdgcn_wmma_f32_16x16x32_bf16(false, (A_), false, (B_), (short)0, (C_), false, false)

static __device__ __forceinline__ bf16 f2bf(float f) {
  unsigned u = __builtin_bit_cast(unsigned, f);
  unsigned r = u + 0x7FFFu + ((u >> 16) & 1u);   // round-to-nearest-even
  unsigned short h = (unsigned short)(r >> 16);
  return __builtin_bit_cast(bf16, h);
}

struct U32x8 { uint4 a, b; };
static __device__ __forceinline__ v16bf make_frag(uint4 lo, uint4 hi) {
  U32x8 t; t.a = lo; t.b = hi;
  return __builtin_bit_cast(v16bf, t);
}
// A frag (row-major LDS tile row): lane pattern k = (e<8?0:16) + mh*8 + (e&7)
static __device__ __forceinline__ v16bf load_a_frag(const bf16* rowptr, int mh) {
  uint4 lo = *(const uint4*)(rowptr + mh * 8);
  uint4 hi = *(const uint4*)(rowptr + 16 + mh * 8);
  return make_frag(lo, hi);
}
// B frag (N-major transposed LDS tile row): k = (e<8?0:16) + mh*8 + (e&7)
static __device__ __forceinline__ v16bf load_b_frag(const bf16* colptr, int mh) {
  uint4 lo = *(const uint4*)(colptr + mh * 8);
  uint4 hi = *(const uint4*)(colptr + 16 + mh * 8);
  return make_frag(lo, hi);
}

// ---- CDNA5 async global->LDS copy (guarded) -------------------------------
typedef int gcc_int4 __attribute__((vector_size(16)));

#if __has_builtin(__builtin_amdgcn_global_load_async_to_lds_b128)
#define ASYNC_CP 1
static __device__ __forceinline__ void cp_async16(const bf16* g, bf16* l) {
  __builtin_amdgcn_global_load_async_to_lds_b128((gcc_int4*)g, (gcc_int4*)l, 0, 0);
}
static __device__ __forceinline__ void cp_async_wait() {
#if __has_builtin(__builtin_amdgcn_s_wait_asynccnt)
  __builtin_amdgcn_s_wait_asynccnt(0);
#else
  asm volatile("s_wait_asynccnt 0x0" ::: "memory");
#endif
}
#else
#define ASYNC_CP 0
#endif

#define BM 128
#define BN 128
#define BK 32

// ---------------------------------------------------------------------------
// Elementwise f32 -> bf16 convert
// ---------------------------------------------------------------------------
__global__ __launch_bounds__(256) void cvt_bf16_kernel(const float* __restrict__ in,
                                                       bf16* __restrict__ out, int n) {
  int i = blockIdx.x * 256 + threadIdx.x;
  if (i < n) out[i] = f2bf(in[i]);
}

// BN fold: scale = g/sqrt(v+eps); shift = b - m*scale   (512 channels)
__global__ __launch_bounds__(256) void bn_prep_kernel(const float* __restrict__ g,
                                                      const float* __restrict__ b,
                                                      const float* __restrict__ m,
                                                      const float* __restrict__ v,
                                                      float* __restrict__ ss) {
  int i = blockIdx.x * 256 + threadIdx.x;
  if (i < 512) {
    float s = g[i] * rsqrtf(v[i] + 1e-5f);
    ss[i]       = s;
    ss[512 + i] = b[i] - m[i] * s;
  }
}

// ---------------------------------------------------------------------------
// Implicit-im2col 3x3 conv (pad 1) + BN + LeakyReLU(0.2)
// ---------------------------------------------------------------------------
__global__ __launch_bounds__(256)
void conv3x3_bn_lrelu_kernel(const bf16* __restrict__ xbf,   // [4,512,64,64]
                             const bf16* __restrict__ wbf,   // [512,4608]
                             const float* __restrict__ bnss, // [2][512]
                             float* __restrict__ outF,       // [4,512,4096]
                             bf16*  __restrict__ outB)       // may be null
{
  __shared__ bf16 As [BM][BK + 8];   // [o][k]   stride 80B
  __shared__ bf16 Bst[BN][BK + 8];   // [tok][k] stride 80B (transposed)

  const int tid  = threadIdx.x;
  const int lane = tid & 31;
  const int wave = tid >> 5;
  const int wr = wave >> 1, wc = wave & 1;
  const int mh = lane >> 4, ml = lane & 15;
  const int m0 = blockIdx.y * BM;
  const int t0 = blockIdx.x * BN;

  // token coords for this thread's fill column
  const int fcol = tid & 127;
  const int fkh  = tid >> 7;          // 0/1 -> k sub-range *16
  const int ftok = t0 + fcol;
  const int fn   = ftok >> 12;
  const int frem = ftok & 4095;
  const int fhh  = frem >> 6, fww = frem & 63;

  v8f acc[2][4];
#pragma unroll
  for (int a = 0; a < 2; ++a)
#pragma unroll
    for (int b = 0; b < 4; ++b)
#pragma unroll
      for (int e = 0; e < 8; ++e) acc[a][b][e] = 0.0f;

  for (int k0 = 0; k0 < 4608; k0 += BK) {
    { // A tile: contiguous weight rows (async DMA to LDS)
      int row = tid >> 1, seg = tid & 1;
      const bf16* src = wbf + (size_t)(m0 + row) * 4608 + k0 + seg * 16;
      bf16* dst = &As[row][seg * 16];
#if ASYNC_CP
      cp_async16(src, dst);
      cp_async16(src + 8, dst + 8);
#else
      const uint4* s4 = (const uint4*)src;
      *(uint4*)(dst + 0) = s4[0];
      *(uint4*)(dst + 8) = s4[1];
#endif
    }
    { // B tile (transposed im2col): unconditional clamped loads + bit-mask zeroing
      union { bf16 h[16]; uint4 q[2]; } tmp;
#pragma unroll
      for (int k = 0; k < 16; ++k) {
        int kg = k0 + fkh * 16 + k;
        int c = kg / 9, r9 = kg - 9 * c;
        int kh = r9 / 3, kw = r9 - 3 * kh;
        int y = fhh + kh - 1, x = fww + kw - 1;
        int yc = min(max(y, 0), 63), xc = min(max(x, 0), 63);
        unsigned short raw = __builtin_bit_cast(
            unsigned short, xbf[(((size_t)fn * 512 + c) << 12) + (yc << 6) + xc]);
        unsigned in = (unsigned)(((unsigned)y < 64u) & ((unsigned)x < 64u));
        unsigned short msk = (unsigned short)(0u - in);   // all-ones if in-range
        tmp.h[k] = __builtin_bit_cast(bf16, (unsigned short)(raw & msk));
      }
      *(uint4*)&Bst[fcol][fkh * 16 + 0] = tmp.q[0];
      *(uint4*)&Bst[fcol][fkh * 16 + 8] = tmp.q[1];
    }
#if ASYNC_CP
    cp_async_wait();
#endif
    __syncthreads();

    v16bf afr[2], bfr[4];
#pragma unroll
    for (int mi = 0; mi < 2; ++mi)
      afr[mi] = load_a_frag(&As[wr * 32 + mi * 16 + ml][0], mh);
#pragma unroll
    for (int ni = 0; ni < 4; ++ni)
      bfr[ni] = load_b_frag(&Bst[wc * 64 + ni * 16 + ml][0], mh);
#pragma unroll
    for (int mi = 0; mi < 2; ++mi)
#pragma unroll
      for (int ni = 0; ni < 4; ++ni)
        acc[mi][ni] = WMMA_BF16(afr[mi], bfr[ni], acc[mi][ni]);
    __syncthreads();
  }

#pragma unroll
  for (int mi = 0; mi < 2; ++mi) {
#pragma unroll
    for (int ni = 0; ni < 4; ++ni) {
      int col = t0 + wc * 64 + ni * 16 + ml;
      int n = col >> 12, rem = col & 4095;
#pragma unroll
      for (int r = 0; r < 8; ++r) {
        int o = m0 + wr * 32 + mi * 16 + mh * 8 + r;
        float v = acc[mi][ni][r] * bnss[o] + bnss[512 + o];
        v = v > 0.0f ? v : 0.2f * v;
        size_t idx = (((size_t)n * 512 + o) << 12) + rem;
        outF[idx] = v;
        if (outB) outB[idx] = f2bf(v);
      }
    }
  }
}

// ---------------------------------------------------------------------------
// Generic 1x1-conv GEMM:  out[n,o,i] = sum_c A[o,c]*B[n,c,i] + bias[o]
// ---------------------------------------------------------------------------
__global__ __launch_bounds__(256)
void gemm1x1_kernel(const bf16* __restrict__ A, const float* __restrict__ bias,
                    const bf16* __restrict__ B, int M,
                    bf16* __restrict__ outB, float* __restrict__ outF)
{
  __shared__ bf16 As [BM][BK + 8];
  __shared__ bf16 Bst[BN][BK + 8];

  const int tid = threadIdx.x, lane = tid & 31, wave = tid >> 5;
  const int wr = wave >> 1, wc = wave & 1, mh = lane >> 4, ml = lane & 15;
  const int m0 = blockIdx.y * BM;
  const int i0 = blockIdx.x * BN;
  const int n  = blockIdx.z;
  const bf16* Bn = B + ((size_t)n * 512 << 12);
  const int fcol = tid & 127, fkh = tid >> 7;

  v8f acc[2][4];
#pragma unroll
  for (int a = 0; a < 2; ++a)
#pragma unroll
    for (int b = 0; b < 4; ++b)
#pragma unroll
      for (int e = 0; e < 8; ++e) acc[a][b][e] = 0.0f;

  for (int k0 = 0; k0 < 512; k0 += BK) {
    { // A tile with M masking: async copy in-range rows, zero-fill the rest
      int row = tid >> 1, seg = tid & 1;
      bf16* dst = &As[row][seg * 16];
      if (m0 + row < M) {
        const bf16* src = A + (size_t)(m0 + row) * 512 + k0 + seg * 16;
#if ASYNC_CP
        cp_async16(src, dst);
        cp_async16(src + 8, dst + 8);
#else
        const uint4* s4 = (const uint4*)src;
        *(uint4*)(dst + 0) = s4[0];
        *(uint4*)(dst + 8) = s4[1];
#endif
      } else {
        uint4 z; z.x = z.y = z.z = z.w = 0u;
        *(uint4*)(dst + 0) = z;
        *(uint4*)(dst + 8) = z;
      }
    }
    { // B tile (transposed): wave-coalesced u16 gathers, vector LDS stores
      union { bf16 h[16]; uint4 q[2]; } tmp;
#pragma unroll
      for (int k = 0; k < 16; ++k)
        tmp.h[k] = Bn[((size_t)(k0 + fkh * 16 + k) << 12) + i0 + fcol];
      *(uint4*)&Bst[fcol][fkh * 16 + 0] = tmp.q[0];
      *(uint4*)&Bst[fcol][fkh * 16 + 8] = tmp.q[1];
    }
#if ASYNC_CP
    cp_async_wait();
#endif
    __syncthreads();

    v16bf afr[2], bfr[4];
#pragma unroll
    for (int mi = 0; mi < 2; ++mi)
      afr[mi] = load_a_frag(&As[wr * 32 + mi * 16 + ml][0], mh);
#pragma unroll
    for (int ni = 0; ni < 4; ++ni)
      bfr[ni] = load_b_frag(&Bst[wc * 64 + ni * 16 + ml][0], mh);
#pragma unroll
    for (int mi = 0; mi < 2; ++mi)
#pragma unroll
      for (int ni = 0; ni < 4; ++ni)
        acc[mi][ni] = WMMA_BF16(afr[mi], bfr[ni], acc[mi][ni]);
    __syncthreads();
  }

#pragma unroll
  for (int mi = 0; mi < 2; ++mi) {
#pragma unroll
    for (int ni = 0; ni < 4; ++ni) {
      int i = i0 + wc * 64 + ni * 16 + ml;
#pragma unroll
      for (int r = 0; r < 8; ++r) {
        int o = m0 + wr * 32 + mi * 16 + mh * 8 + r;
        if (o < M) {
          float v = acc[mi][ni][r] + bias[o];
          size_t idx = (((size_t)n * M + o) << 12) + i;
          if (outB) outB[idx] = f2bf(v);
          if (outF) outF[idx] = v;
        }
      }
    }
  }
}

// ---------------------------------------------------------------------------
// Attention scores: attn[n,i,j] = sum_c q[n,c,i]*k[n,c,j]   (K=64)
// ---------------------------------------------------------------------------
__global__ __launch_bounds__(256)
void gemm_scores_kernel(const bf16* __restrict__ q, const bf16* __restrict__ kmat,
                        float* __restrict__ attn)
{
  __shared__ bf16 As [BM][64 + 8];   // [i][c]  stride 144B
  __shared__ bf16 Bst[BN][64 + 8];   // [j][c]  stride 144B

  const int tid = threadIdx.x, lane = tid & 31, wave = tid >> 5;
  const int wr = wave >> 1, wc = wave & 1, mh = lane >> 4, ml = lane & 15;
  const int i0 = blockIdx.y * BM;
  const int j0 = blockIdx.x * BN;
  const int n  = blockIdx.z;
  const bf16* qn = q    + ((size_t)n * 64 << 12);
  const bf16* kn = kmat + ((size_t)n * 64 << 12);
  const int frow = tid & 127, fhalf = tid >> 7;   // 32 c's per thread

  { // A = q^T  (wave-coalesced u16 gathers)
    union { bf16 h[32]; uint4 q4[4]; } tmp;
#pragma unroll
    for (int cc = 0; cc < 32; ++cc)
      tmp.h[cc] = qn[((size_t)(fhalf * 32 + cc) << 12) + i0 + frow];
#pragma unroll
    for (int s = 0; s < 4; ++s) *(uint4*)&As[frow][fhalf * 32 + s * 8] = tmp.q4[s];
  }
  { // B = k (transposed into [j][c])
    union { bf16 h[32]; uint4 q4[4]; } tmp;
#pragma unroll
    for (int cc = 0; cc < 32; ++cc)
      tmp.h[cc] = kn[((size_t)(fhalf * 32 + cc) << 12) + j0 + frow];
#pragma unroll
    for (int s = 0; s < 4; ++s) *(uint4*)&Bst[frow][fhalf * 32 + s * 8] = tmp.q4[s];
  }
  __syncthreads();

  v8f acc[2][4];
#pragma unroll
  for (int a = 0; a < 2; ++a)
#pragma unroll
    for (int b = 0; b < 4; ++b)
#pragma unroll
      for (int e = 0; e < 8; ++e) acc[a][b][e] = 0.0f;

#pragma unroll
  for (int ks = 0; ks < 2; ++ks) {
    v16bf afr[2], bfr[4];
#pragma unroll
    for (int mi = 0; mi < 2; ++mi)
      afr[mi] = load_a_frag(&As[wr * 32 + mi * 16 + ml][ks * 32], mh);
#pragma unroll
    for (int ni = 0; ni < 4; ++ni)
      bfr[ni] = load_b_frag(&Bst[wc * 64 + ni * 16 + ml][ks * 32], mh);
#pragma unroll
    for (int mi = 0; mi < 2; ++mi)
#pragma unroll
      for (int ni = 0; ni < 4; ++ni)
        acc[mi][ni] = WMMA_BF16(afr[mi], bfr[ni], acc[mi][ni]);
  }

#pragma unroll
  for (int mi = 0; mi < 2; ++mi)
#pragma unroll
    for (int ni = 0; ni < 4; ++ni) {
      int j = j0 + wc * 64 + ni * 16 + ml;
#pragma unroll
      for (int r = 0; r < 8; ++r) {
        int i = i0 + wr * 32 + mi * 16 + mh * 8 + r;
        attn[((size_t)n << 24) + ((size_t)i << 12) + j] = acc[mi][ni][r];
      }
    }
}

// Row softmax over 4096 columns, in place. Grid: 4*4096 rows.
__global__ __launch_bounds__(256)
void softmax_rows_kernel(float* __restrict__ attn)
{
  __shared__ float red[256];
  float* p = attn + ((size_t)blockIdx.x << 12);
  int tid = threadIdx.x;
  float mx = -1e30f;
  for (int i = tid; i < 4096; i += 256) mx = fmaxf(mx, p[i]);
  red[tid] = mx; __syncthreads();
  for (int s = 128; s > 0; s >>= 1) { if (tid < s) red[tid] = fmaxf(red[tid], red[tid + s]); __syncthreads(); }
  mx = red[0]; __syncthreads();
  float sum = 0.f;
  for (int i = tid; i < 4096; i += 256) { float e = __expf(p[i] - mx); p[i] = e; sum += e; }
  red[tid] = sum; __syncthreads();
  for (int s = 128; s > 0; s >>= 1) { if (tid < s) red[tid] += red[tid + s]; __syncthreads(); }
  float inv = 1.0f / red[0];
  __syncthreads();
  for (int i = tid; i < 4096; i += 256) p[i] *= inv;
}

// ---------------------------------------------------------------------------
// out[n,c,i] = alpha * (sum_j v[n,c,j]*attn[n,i,j]) + feat[n,c,i]  -> bf16
// ---------------------------------------------------------------------------
__global__ __launch_bounds__(256)
void gemm_av_kernel(const bf16* __restrict__ vmat, const float* __restrict__ attn,
                    const float* __restrict__ alpha, const float* __restrict__ featF,
                    bf16* __restrict__ outB)
{
  __shared__ bf16 As [BM][BK + 8];
  __shared__ bf16 Bst[BN][BK + 8];

  const int tid = threadIdx.x, lane = tid & 31, wave = tid >> 5;
  const int wr = wave >> 1, wc = wave & 1, mh = lane >> 4, ml = lane & 15;
  const int c0 = blockIdx.y * BM;
  const int i0 = blockIdx.x * BN;
  const int n  = blockIdx.z;
  const bf16*  vn = vmat + ((size_t)n * 512 << 12);
  const float* an = attn + ((size_t)n << 24);
  const int fcol = tid & 127, fkh = tid >> 7;

  v8f acc[2][4];
#pragma unroll
  for (int a = 0; a < 2; ++a)
#pragma unroll
    for (int b = 0; b < 4; ++b)
#pragma unroll
      for (int e = 0; e < 8; ++e) acc[a][b][e] = 0.0f;

  for (int j0 = 0; j0 < 4096; j0 += BK) {
    { // A = v tile (contiguous, async DMA to LDS)
      int row = tid >> 1, seg = tid & 1;
      const bf16* src = vn + ((size_t)(c0 + row) << 12) + j0 + seg * 16;
      bf16* dst = &As[row][seg * 16];
#if ASYNC_CP
      cp_async16(src, dst);
      cp_async16(src + 8, dst + 8);
#else
      const uint4* s4 = (const uint4*)src;
      *(uint4*)(dst + 0) = s4[0];
      *(uint4*)(dst + 8) = s4[1];
#endif
    }
    { // B[i][j] = attn row segment: contiguous 64B f32 reads, cvt, transpose-store
      const float4* src = (const float4*)(an + ((size_t)(i0 + fcol) << 12) + j0 + fkh * 16);
      union { bf16 h[16]; uint4 q[2]; } tmp;
#pragma unroll
      for (int s = 0; s < 4; ++s) {
        float4 f = src[s];
        tmp.h[s * 4 + 0] = f2bf(f.x);
        tmp.h[s * 4 + 1] = f2bf(f.y);
        tmp.h[s * 4 + 2] = f2bf(f.z);
        tmp.h[s * 4 + 3] = f2bf(f.w);
      }
      *(uint4*)&Bst[fcol][fkh * 16 + 0] = tmp.q[0];
      *(uint4*)&Bst[fcol][fkh * 16 + 8] = tmp.q[1];
    }
#if ASYNC_CP
    cp_async_wait();
#endif
    __syncthreads();

    v16bf afr[2], bfr[4];
#pragma unroll
    for (int mi = 0; mi < 2; ++mi)
      afr[mi] = load_a_frag(&As[wr * 32 + mi * 16 + ml][0], mh);
#pragma unroll
    for (int ni = 0; ni < 4; ++ni)
      bfr[ni] = load_b_frag(&Bst[wc * 64 + ni * 16 + ml][0], mh);
#pragma unroll
    for (int mi = 0; mi < 2; ++mi)
#pragma unroll
      for (int ni = 0; ni < 4; ++ni)
        acc[mi][ni] = WMMA_BF16(afr[mi], bfr[ni], acc[mi][ni]);
    __syncthreads();
  }

  float al = alpha[0];
#pragma unroll
  for (int mi = 0; mi < 2; ++mi)
#pragma unroll
    for (int ni = 0; ni < 4; ++ni) {
      int i = i0 + wc * 64 + ni * 16 + ml;
#pragma unroll
      for (int r = 0; r < 8; ++r) {
        int c = c0 + wr * 32 + mi * 16 + mh * 8 + r;
        size_t idx = (((size_t)n * 512 + c) << 12) + i;
        outB[idx] = f2bf(al * acc[mi][ni][r] + featF[idx]);
      }
    }
}

// ---------------------------------------------------------------------------
// CAM small kernels
// ---------------------------------------------------------------------------
__global__ __launch_bounds__(256)
void chan_mean_kernel(const float* __restrict__ f, float* __restrict__ s)
{
  __shared__ float red[256];
  size_t base = (((size_t)blockIdx.y * 512 + blockIdx.x) << 12);
  int tid = threadIdx.x;
  float sum = 0.f;
  for (int i = tid; i < 4096; i += 256) sum += f[base + i];
  red[tid] = sum; __syncthreads();
  for (int st = 128; st > 0; st >>= 1) { if (tid < st) red[tid] += red[tid + st]; __syncthreads(); }
  if (tid == 0) s[blockIdx.y * 512 + blockIdx.x] = red[0] * (1.0f / 4096.0f);
}

__global__ __launch_bounds__(512)
void cam_mlp_kernel(const float* __restrict__ s,
                    const float* __restrict__ Wc1, const float* __restrict__ bc1,
                    const float* __restrict__ Wc2, const float* __restrict__ bc2,
                    float* __restrict__ sc)
{
  __shared__ float sh_s[512];
  __shared__ float sh_h[64];
  int n = blockIdx.x, tid = threadIdx.x;
  sh_s[tid] = s[n * 512 + tid];
  __syncthreads();
  if (tid < 64) {
    float a = bc1[tid];
    for (int c = 0; c < 512; ++c) a += Wc1[tid * 512 + c] * sh_s[c];
    sh_h[tid] = a > 0.f ? a : 0.2f * a;
  }
  __syncthreads();
  float a = bc2[tid];
  for (int c = 0; c < 64; ++c) a += Wc2[tid * 64 + c] * sh_h[c];
  sc[n * 512 + tid] = 1.0f / (1.0f + __expf(-a));
}

__global__ __launch_bounds__(256)
void cam_scale_kernel(const float* __restrict__ f, const float* __restrict__ sc,
                      bf16* __restrict__ out, int total)
{
  int i = blockIdx.x * 256 + threadIdx.x;
  if (i < total) out[i] = f2bf(f[i] * sc[i >> 12]);
}

// ---------------------------------------------------------------------------
// Bilinear x2 upsample (half-pixel) of both branches + add -> d_out
// ---------------------------------------------------------------------------
__global__ __launch_bounds__(256)
void upsample_add_kernel(const float* __restrict__ pa, const float* __restrict__ ca,
                         float* __restrict__ out)
{
  int idx = blockIdx.x * 256 + threadIdx.x;
  if (idx >= 4 * 64 * 128 * 128) return;
  int X = idx & 127, Y = (idx >> 7) & 127;
  int ch = idx >> 14;   // n*64 + o
  float sy = Y * 0.5f - 0.25f, sx = X * 0.5f - 0.25f;
  int y0 = (int)floorf(sy), x0 = (int)floorf(sx);
  float fy = sy - (float)y0, fx = sx - (float)x0;
  int y1 = y0 + 1, x1 = x0 + 1;
  y0 = min(max(y0, 0), 63); y1 = min(max(y1, 0), 63);
  x0 = min(max(x0, 0), 63); x1 = min(max(x1, 0), 63);
  float w00 = (1.f - fy) * (1.f - fx), w01 = (1.f - fy) * fx;
  float w10 = fy * (1.f - fx),         w11 = fy * fx;
  size_t base = ((size_t)ch) << 12;
  float p = pa[base + (y0 << 6) + x0] * w00 + pa[base + (y0 << 6) + x1] * w01 +
            pa[base + (y1 << 6) + x0] * w10 + pa[base + (y1 << 6) + x1] * w11;
  float c = ca[base + (y0 << 6) + x0] * w00 + ca[base + (y0 << 6) + x1] * w01 +
            ca[base + (y1 << 6) + x0] * w10 + ca[base + (y1 << 6) + x1] * w11;
  out[idx] = p + c;
}

// ---------------------------------------------------------------------------
// Launch
// ---------------------------------------------------------------------------
extern "C" void kernel_launch(void* const* d_in, const int* in_sizes, int n_in,
                              void* d_out, int out_size, void* d_ws, size_t ws_size,
                              hipStream_t stream) {
  (void)in_sizes; (void)n_in; (void)out_size; (void)ws_size;
  const float* x     = (const float*)d_in[0];
  const float* Wpam  = (const float*)d_in[1];
  const float* pg    = (const float*)d_in[2];
  const float* pb    = (const float*)d_in[3];
  const float* pm    = (const float*)d_in[4];
  const float* pv    = (const float*)d_in[5];
  const float* Wq    = (const float*)d_in[6];
  const float* bq    = (const float*)d_in[7];
  const float* Wk    = (const float*)d_in[8];
  const float* bk    = (const float*)d_in[9];
  const float* Wv    = (const float*)d_in[10];
  const float* bv    = (const float*)d_in[11];
  const float* alpha = (const float*)d_in[12];
  const float* Wpo   = (const float*)d_in[13];
  const float* bpo   = (const float*)d_in[14];
  const float* Wcam  = (const float*)d_in[15];
  const float* cg    = (const float*)d_in[16];
  const float* cb    = (const float*)d_in[17];
  const float* cm    = (const float*)d_in[18];
  const float* cv    = (const float*)d_in[19];
  const float* Wc1   = (const float*)d_in[20];
  const float* bc1   = (const float*)d_in[21];
  const float* Wc2   = (const float*)d_in[22];
  const float* bc2   = (const float*)d_in[23];
  const float* Wco   = (const float*)d_in[24];
  const float* bco   = (const float*)d_in[25];
  float* out = (float*)d_out;

  char* ws = (char*)d_ws;
  size_t off = 0;
  auto take = [&](size_t bytes) -> void* {
    void* p = ws + off;
    off = (off + bytes + 255) & ~(size_t)255;
    return p;
  };

  bf16*  xbf      = (bf16*) take((size_t)8388608 * 2);
  bf16*  wpamB    = (bf16*) take((size_t)2359296 * 2);
  bf16*  wcamB    = (bf16*) take((size_t)2359296 * 2);
  bf16*  wqB      = (bf16*) take(32768 * 2);
  bf16*  wkB      = (bf16*) take(32768 * 2);
  bf16*  wvB      = (bf16*) take(262144 * 2);
  bf16*  wpoB     = (bf16*) take(32768 * 2);
  bf16*  wcoB     = (bf16*) take(32768 * 2);
  float* bnPam    = (float*)take(1024 * 4);
  float* bnCam    = (float*)take(1024 * 4);
  float* featPamF = (float*)take((size_t)8388608 * 4);
  bf16*  featPamB = (bf16*) take((size_t)8388608 * 2);
  float* featCamF = (float*)take((size_t)8388608 * 4);
  bf16*  qB       = (bf16*) take((size_t)1048576 * 2);
  bf16*  kB       = (bf16*) take((size_t)1048576 * 2);
  bf16*  vB       = (bf16*) take((size_t)8388608 * 2);
  float* attn     = (float*)take((size_t)67108864 * 4);
  bf16*  pamA     = (bf16*) take((size_t)8388608 * 2);
  bf16*  camA     = (bf16*) take((size_t)8388608 * 2);
  float* sMean    = (float*)take(2048 * 4);
  float* sScale   = (float*)take(2048 * 4);
  float* pamOut   = (float*)take((size_t)1048576 * 4);
  float* camOut   = (float*)take((size_t)1048576 * 4);

  cvt_bf16_kernel<<<(8388608 + 255) / 256, 256, 0, stream>>>(x, xbf, 8388608);
  cvt_bf16_kernel<<<(2359296 + 255) / 256, 256, 0, stream>>>(Wpam, wpamB, 2359296);
  cvt_bf16_kernel<<<(2359296 + 255) / 256, 256, 0, stream>>>(Wcam, wcamB, 2359296);
  cvt_bf16_kernel<<<(32768 + 255) / 256, 256, 0, stream>>>(Wq, wqB, 32768);
  cvt_bf16_kernel<<<(32768 + 255) / 256, 256, 0, stream>>>(Wk, wkB, 32768);
  cvt_bf16_kernel<<<(262144 + 255) / 256, 256, 0, stream>>>(Wv, wvB, 262144);
  cvt_bf16_kernel<<<(32768 + 255) / 256, 256, 0, stream>>>(Wpo, wpoB, 32768);
  cvt_bf16_kernel<<<(32768 + 255) / 256, 256, 0, stream>>>(Wco, wcoB, 32768);
  bn_prep_kernel<<<2, 256, 0, stream>>>(pg, pb, pm, pv, bnPam);
  bn_prep_kernel<<<2, 256, 0, stream>>>(cg, cb, cm, cv, bnCam);

  conv3x3_bn_lrelu_kernel<<<dim3(128, 4), 256, 0, stream>>>(xbf, wpamB, bnPam, featPamF, featPamB);
  conv3x3_bn_lrelu_kernel<<<dim3(128, 4), 256, 0, stream>>>(xbf, wcamB, bnCam, featCamF, nullptr);

  gemm1x1_kernel<<<dim3(32, 1, 4), 256, 0, stream>>>(wqB, bq, featPamB, 64, qB, nullptr);
  gemm1x1_kernel<<<dim3(32, 1, 4), 256, 0, stream>>>(wkB, bk, featPamB, 64, kB, nullptr);
  gemm1x1_kernel<<<dim3(32, 4, 4), 256, 0, stream>>>(wvB, bv, featPamB, 512, vB, nullptr);

  gemm_scores_kernel<<<dim3(32, 32, 4), 256, 0, stream>>>(qB, kB, attn);
  softmax_rows_kernel<<<16384, 256, 0, stream>>>(attn);
  gemm_av_kernel<<<dim3(32, 4, 4), 256, 0, stream>>>(vB, attn, alpha, featPamF, pamA);

  gemm1x1_kernel<<<dim3(32, 1, 4), 256, 0, stream>>>(wpoB, bpo, pamA, 64, nullptr, pamOut);

  chan_mean_kernel<<<dim3(512, 4), 256, 0, stream>>>(featCamF, sMean);
  cam_mlp_kernel<<<4, 512, 0, stream>>>(sMean, Wc1, bc1, Wc2, bc2, sScale);
  cam_scale_kernel<<<(8388608 + 255) / 256, 256, 0, stream>>>(featCamF, sScale, camA, 8388608);
  gemm1x1_kernel<<<dim3(32, 1, 4), 256, 0, stream>>>(wcoB, bco, camA, 64, nullptr, camOut);

  upsample_add_kernel<<<16384, 256, 0, stream>>>(pamOut, camOut, out);
}
